// CompressionTransformerLayer_86861418594518
// MI455X (gfx1250) — compile-verified, matrix operand order checked by reading, not run on hardware
//
#include <hip/hip_runtime.h>
#include <hip/hip_bf16.h>
#include <math.h>
#include <stdint.h>

// ---------------------------------------------------------------------------
// CDNA5 (gfx1250) transformer decoder layer:
//  - all GEMMs / attention matmuls via v_wmma_f32_16x16x32_f16 (f32 accumulate)
//  - tile staging via global_load_async_to_lds_b128 (ASYNCcnt), double-buffered
//  - V matrices produced transposed so every WMMA fragment is a contiguous
//    (ds_load_b128-friendly) row-major gather
// ---------------------------------------------------------------------------

typedef _Float16 h16;
typedef __attribute__((ext_vector_type(16))) _Float16 v16h;
typedef __attribute__((ext_vector_type(8)))  float    v8f;

#define D_MODEL 1024
#define NHEAD   16
#define HDIM    64
#define DFF_    4096
#define BB      4
#define QL      256
#define SL      8192

// ---- CDNA5 async copy: global memory -> LDS, 16 bytes per lane ------------
__device__ __forceinline__ void async_load_b128(uint32_t lds_off, const void* gptr) {
  asm volatile("global_load_async_to_lds_b128 %0, %1, off"
               :: "v"(lds_off), "v"(gptr)
               : "memory");
}
__device__ __forceinline__ void wait_async_lds() {
  asm volatile("s_wait_asynccnt 0x0" ::: "memory");
}
__device__ __forceinline__ uint32_t lds_off_of(const void* shared_ptr) {
  return (uint32_t)(uintptr_t)shared_ptr;
}

// ---- WMMA fragment helper (ISA 7.12.2 layouts, wave32) --------------------
__device__ __forceinline__ int kmap(int v, int hlf) {
  return (v < 4 ? 2 * v : 16 + 2 * (v - 4)) + 8 * hlf;
}

// element(row, k) = lds[(row0+row)*stride + k0 + k]; lane carries `row`
// (used for A fragments with row=M, and B fragments with row=N when the LDS
//  tile is stored [N][K]-major -- k-pairs contiguous => ds_load_b128 pairs)
__device__ __forceinline__ v16h frag_rowmajor(const h16* lds, int stride, int row0, int k0, int lane) {
  int hlf = lane >> 4, r = lane & 15;
  const h16* base = lds + (size_t)(row0 + r) * stride + k0;
  v16h f;
#pragma unroll
  for (int v = 0; v < 8; ++v) {
    int kk = kmap(v, hlf);
    f[2 * v]     = base[kk];
    f[2 * v + 1] = base[kk + 1];
  }
  return f;
}

// ---------------------------------------------------------------------------
// f32 -> f16 conversion (weights / context)
// ---------------------------------------------------------------------------
__global__ void f32_to_f16_kernel(const float* __restrict__ in, h16* __restrict__ out, size_t n) {
  size_t i = (size_t)blockIdx.x * blockDim.x + threadIdx.x;
  size_t stride = (size_t)gridDim.x * blockDim.x;
  for (; i < n; i += stride) out[i] = (h16)in[i];
}

// ---------------------------------------------------------------------------
// LayerNorm over last dim (D=1024), one block per row, f16 output
// ---------------------------------------------------------------------------
__global__ __launch_bounds__(256) void ln_f16_kernel(const float* __restrict__ x,
                                                     const float* __restrict__ g,
                                                     const float* __restrict__ b,
                                                     h16* __restrict__ out, int D) {
  int row = blockIdx.x, tid = threadIdx.x;
  const float* xr = x + (size_t)row * D;
  __shared__ float red[256];
  __shared__ float mu_s, rs_s;

  float s = 0.f;
  for (int i = tid; i < D; i += 256) s += xr[i];
  red[tid] = s; __syncthreads();
  for (int off = 128; off > 0; off >>= 1) {
    if (tid < off) red[tid] += red[tid + off];
    __syncthreads();
  }
  if (tid == 0) mu_s = red[0] / (float)D;
  __syncthreads();
  float mu = mu_s;

  float s2 = 0.f;
  for (int i = tid; i < D; i += 256) { float d = xr[i] - mu; s2 += d * d; }
  red[tid] = s2; __syncthreads();
  for (int off = 128; off > 0; off >>= 1) {
    if (tid < off) red[tid] += red[tid + off];
    __syncthreads();
  }
  if (tid == 0) rs_s = rsqrtf(red[0] / (float)D + 1e-5f);
  __syncthreads();
  float rs = rs_s;

  h16* orow = out + (size_t)row * D;
  for (int i = tid; i < D; i += 256)
    orow[i] = (h16)((xr[i] - mu) * rs * g[i] + b[i]);
}

// ---------------------------------------------------------------------------
// Tiled WMMA GEMM:  C[M,N] = A[M,K] (f16) * W[N,K]^T (f16)  + bias (+gelu)(+residual)
// Block tile 128x128, BK=32, 256 threads = 8 waves arranged 4(M) x 2(N),
// wave tile 32x64 = 2x4 fragments of 16x16.  8 WMMAs / wave / k-step.
// Async b128 staging, double-buffered.  Epilogue variants are compile-time.
// O16T writes the f16 result transposed: out[col*M + row] (contiguous in row).
// M%128==N%128==K%32==0 assumed.
// ---------------------------------------------------------------------------
#define GBM 128
#define GBN 128
#define GBK 32
#define GLD 40   // padded LDS stride (f16 elems); 80 B per row (16B aligned)

template <int DOGELU, int HASRES, int O32, int O16, int O16T>
__global__ __launch_bounds__(256) void gemm_wmma_f16(
    const h16* __restrict__ A, const h16* __restrict__ W,
    const float* __restrict__ bias, const float* __restrict__ residual,
    float* __restrict__ outF32, h16* __restrict__ outF16,
    int M, int N, int K) {
  __shared__ h16 As[2][GBM * GLD];
  __shared__ h16 Bs[2][GBN * GLD];

  int tid = threadIdx.x;
  int lane = tid & 31;
  int w = tid >> 5;
  int wy = w >> 1, wx = w & 1;            // 4 x 2 wave grid
  int bm0 = blockIdx.y * GBM;
  int bn0 = blockIdx.x * GBN;

  uint32_t asB = lds_off_of(&As[0][0]);
  uint32_t bsB = lds_off_of(&Bs[0][0]);
  const uint32_t bufBytes = GBM * GLD * 2;

  auto issue_tile = [&](int kt, int buf) {
    int kb = kt * GBK;
#pragma unroll
    for (int i = 0; i < 2; ++i) {
      int c = i * 256 + tid;
      int r = c >> 2, k8 = (c & 3) * 8;              // 4 chunks of 8 f16 per row
      async_load_b128(asB + (uint32_t)buf * bufBytes + (uint32_t)(r * GLD + k8) * 2,
                      &A[(size_t)(bm0 + r) * K + kb + k8]);
      async_load_b128(bsB + (uint32_t)buf * bufBytes + (uint32_t)(r * GLD + k8) * 2,
                      &W[(size_t)(bn0 + r) * K + kb + k8]);
    }
  };

  v8f zero = {0.f, 0.f, 0.f, 0.f, 0.f, 0.f, 0.f, 0.f};
  v8f acc[2][4];
#pragma unroll
  for (int i = 0; i < 2; ++i)
#pragma unroll
    for (int j = 0; j < 4; ++j) acc[i][j] = zero;

  int nkt = K / GBK;
  issue_tile(0, 0);
  wait_async_lds();
  __syncthreads();

  for (int kt = 0; kt < nkt; ++kt) {
    int cur = kt & 1;
    if (kt + 1 < nkt) issue_tile(kt + 1, cur ^ 1);   // overlap copy with compute

    const h16* Ac = As[cur];
    const h16* Bc = Bs[cur];
    v16h af[2], bf[4];
#pragma unroll
    for (int i = 0; i < 2; ++i) af[i] = frag_rowmajor(Ac, GLD, wy * 32 + i * 16, 0, lane);
    // logical B(k,n) = W[n][k]; Bs is [n][k]-major -> row-major gather over n
#pragma unroll
    for (int j = 0; j < 4; ++j) bf[j] = frag_rowmajor(Bc, GLD, wx * 64 + j * 16, 0, lane);
#pragma unroll
    for (int i = 0; i < 2; ++i)
#pragma unroll
      for (int j = 0; j < 4; ++j)
        acc[i][j] = __builtin_amdgcn_wmma_f32_16x16x32_f16(false, af[i], false, bf[j],
                                                           (short)0, acc[i][j], false, false);
    wait_async_lds();
    __syncthreads();
  }

  int hlf = lane >> 4, cix = lane & 15;
#pragma unroll
  for (int i = 0; i < 2; ++i) {
#pragma unroll
    for (int j = 0; j < 4; ++j) {
      int col = bn0 + wx * 64 + j * 16 + cix;
      float bval = bias[col];
#pragma unroll
      for (int r = 0; r < 8; ++r) {
        int row = bm0 + wy * 32 + i * 16 + r + 8 * hlf;
        float v = acc[i][j][r] + bval;
        if (DOGELU) v = 0.5f * v * (1.0f + erff(v * 0.70710678118654752f));
        if (HASRES) v += residual[(size_t)row * N + col];
        if (O32)  outF32[(size_t)row * N + col] = v;
        if (O16)  outF16[(size_t)row * N + col] = (h16)v;
        if (O16T) outF16[(size_t)col * M + row] = (h16)v;  // contiguous along r
      }
    }
  }
}

// ---------------------------------------------------------------------------
// Streaming (flash-style) multi-head attention with WMMA for QK^T and P*V.
// V is provided TRANSPOSED ([head-dim][tokens], leading dim vTld) so both
// WMMA B-operands are contiguous row-major gathers in LDS.
// Block = 64 query rows for one (b,h); loops over Lk in 64-wide chunks with
// online softmax (4 threads per row, __shfl_xor segment reductions).
// K/V chunks double-buffered through the async engine.
// 256 threads = 8 waves as 4(M=16 rows each) x 2(N=32 cols each).
// ---------------------------------------------------------------------------
#define ATT_LDH 72   // f16 LDS stride (144 B per row, 16B aligned)
#define ATT_LDS 68   // f32 score stride

__global__ __launch_bounds__(256) void attn_wmma(
    const h16* __restrict__ Qg, const h16* __restrict__ Kg, const h16* __restrict__ VTg,
    h16* __restrict__ Og,
    int Lq, int Lk, int qStride, int kStride, int vTld,
    int qco, int kco) {
  __shared__ h16 Qs[64 * ATT_LDH];
  __shared__ h16 Ks[2][64 * ATT_LDH];
  __shared__ h16 Vs[2][64 * ATT_LDH];   // transposed: [head-dim][s]
  __shared__ h16 Ps[64 * ATT_LDH];
  __shared__ float Ss[64 * ATT_LDS];
  __shared__ float mrow[64], lrow[64], arow[64];

  int tid = threadIdx.x, lane = tid & 31, w = tid >> 5;
  int wy = w >> 1, wx = w & 1;
  int hlf = lane >> 4, cix = lane & 15;

  int qTiles = Lq / 64;
  int qt = blockIdx.x % qTiles;
  int bh = blockIdx.x / qTiles;
  int b = bh / NHEAD, h = bh % NHEAD;

  const h16* qb  = Qg  + (size_t)b * Lq * qStride + h * HDIM + qco;
  const h16* kb  = Kg  + (size_t)b * Lk * kStride + h * HDIM + kco;
  const h16* vtb = VTg + (size_t)(h * HDIM) * vTld + (size_t)b * Lk;  // [d][s]
  h16* ob = Og + (size_t)b * Lq * D_MODEL + h * HDIM;

  uint32_t qsB = lds_off_of(Qs);
  uint32_t ksB = lds_off_of(&Ks[0][0]);
  uint32_t vsB = lds_off_of(&Vs[0][0]);
  const uint32_t kvBytes = 64 * ATT_LDH * 2;

  auto issue_kv = [&](int s0, int buf) {
#pragma unroll
    for (int i = 0; i < 2; ++i) {
      int c = i * 256 + tid;
      int r = c >> 3, d8 = (c & 7) * 8;
      // K chunk: rows = tokens s, cols = head-dim
      async_load_b128(ksB + (uint32_t)buf * kvBytes + (uint32_t)(r * ATT_LDH + d8) * 2,
                      &kb[(size_t)(s0 + r) * kStride + d8]);
      // V^T chunk: rows = head-dim d, cols = tokens s
      async_load_b128(vsB + (uint32_t)buf * kvBytes + (uint32_t)(r * ATT_LDH + d8) * 2,
                      &vtb[(size_t)r * vTld + s0 + d8]);
    }
  };

  // Q tile (64 rows x 64 f16) + first K/V chunk
#pragma unroll
  for (int i = 0; i < 2; ++i) {
    int c = i * 256 + tid;
    int r = c >> 3, d8 = (c & 7) * 8;
    async_load_b128(qsB + (uint32_t)(r * ATT_LDH + d8) * 2,
                    &qb[(size_t)(qt * 64 + r) * qStride + d8]);
  }
  issue_kv(0, 0);
  if (tid < 64) { mrow[tid] = -3.0e38f; lrow[tid] = 0.0f; }
  v8f zero = {0.f, 0.f, 0.f, 0.f, 0.f, 0.f, 0.f, 0.f};
  v8f oacc[2]; oacc[0] = zero; oacc[1] = zero;
  wait_async_lds();
  __syncthreads();

  int nCh = Lk / 64;
  for (int ch = 0; ch < nCh; ++ch) {
    int cur = ch & 1;
    if (ch + 1 < nCh) issue_kv((ch + 1) * 64, cur ^ 1);   // overlap with compute
    const h16* Kc = Ks[cur];
    const h16* Vc = Vs[cur];

    // scores: Q(64x64) * K^T -> 64x64, K-dim = HDIM in two 32-steps
    v8f sacc[2]; sacc[0] = zero; sacc[1] = zero;
#pragma unroll
    for (int ks = 0; ks < 2; ++ks) {
      v16h aq  = frag_rowmajor(Qs, ATT_LDH, wy * 16,       ks * 32, lane);
      // B(k=d, n=s) = K[s][d]: lane carries s, k-pairs contiguous along d
      v16h bk0 = frag_rowmajor(Kc, ATT_LDH, wx * 32,       ks * 32, lane);
      v16h bk1 = frag_rowmajor(Kc, ATT_LDH, wx * 32 + 16,  ks * 32, lane);
      sacc[0] = __builtin_amdgcn_wmma_f32_16x16x32_f16(false, aq, false, bk0, (short)0, sacc[0], false, false);
      sacc[1] = __builtin_amdgcn_wmma_f32_16x16x32_f16(false, aq, false, bk1, (short)0, sacc[1], false, false);
    }
#pragma unroll
    for (int j = 0; j < 2; ++j)
#pragma unroll
      for (int r = 0; r < 8; ++r)
        Ss[(wy * 16 + r + 8 * hlf) * ATT_LDS + wx * 32 + j * 16 + cix] = sacc[j][r] * 0.125f;
    __syncthreads();

    // online softmax: 4 threads per row, 16 columns each; segment reductions
    {
      int row = tid >> 2, sub = tid & 3;
      const float* srow = &Ss[row * ATT_LDS + sub * 16];
      float cmax = -3.0e38f;
#pragma unroll
      for (int c2 = 0; c2 < 16; ++c2) cmax = fmaxf(cmax, srow[c2]);
      cmax = fmaxf(cmax, __shfl_xor(cmax, 1, 4));
      cmax = fmaxf(cmax, __shfl_xor(cmax, 2, 4));
      float m0 = mrow[row];
      float mnew = fmaxf(m0, cmax);
      float al = expf(m0 - mnew);
      float s = 0.f;
      h16* prow = &Ps[row * ATT_LDH + sub * 16];
#pragma unroll
      for (int c2 = 0; c2 < 16; ++c2) {
        float pv = expf(srow[c2] - mnew);
        prow[c2] = (h16)pv;
        s += pv;
      }
      s += __shfl_xor(s, 1, 4);
      s += __shfl_xor(s, 2, 4);
      if (sub == 0) {
        lrow[row] = lrow[row] * al + s;
        mrow[row] = mnew;
        arow[row] = al;
      }
    }
    __syncthreads();

    // rescale accumulators, then O += P(64x64) * V(64x64)
#pragma unroll
    for (int j = 0; j < 2; ++j)
#pragma unroll
      for (int r = 0; r < 8; ++r)
        oacc[j][r] = oacc[j][r] * arow[wy * 16 + r + 8 * hlf];
#pragma unroll
    for (int ks = 0; ks < 2; ++ks) {
      v16h ap  = frag_rowmajor(Ps, ATT_LDH, wy * 16, ks * 32, lane);
      // B(k=s, n=d) = V^T[d][s]: lane carries d, k-pairs contiguous along s
      v16h bv0 = frag_rowmajor(Vc, ATT_LDH, wx * 32,      ks * 32, lane);
      v16h bv1 = frag_rowmajor(Vc, ATT_LDH, wx * 32 + 16, ks * 32, lane);
      oacc[0] = __builtin_amdgcn_wmma_f32_16x16x32_f16(false, ap, false, bv0, (short)0, oacc[0], false, false);
      oacc[1] = __builtin_amdgcn_wmma_f32_16x16x32_f16(false, ap, false, bv1, (short)0, oacc[1], false, false);
    }
    wait_async_lds();
    __syncthreads();
  }

  // normalize + write merged-head output (f16)
#pragma unroll
  for (int j = 0; j < 2; ++j)
#pragma unroll
    for (int r = 0; r < 8; ++r) {
      int row = wy * 16 + r + 8 * hlf;
      float val = oacc[j][r] / lrow[row];
      ob[(size_t)(qt * 64 + row) * D_MODEL + wx * 32 + j * 16 + cix] = (h16)val;
    }
}

// ---------------------------------------------------------------------------
// Host orchestration
// ---------------------------------------------------------------------------
extern "C" void kernel_launch(void* const* d_in, const int* in_sizes, int n_in,
                              void* d_out, int out_size, void* d_ws, size_t ws_size,
                              hipStream_t stream) {
  (void)in_sizes; (void)n_in; (void)out_size; (void)ws_size;

  const float* queries  = (const float*)d_in[0];
  const float* context  = (const float*)d_in[1];
  const float* sa_in_w  = (const float*)d_in[2];
  const float* sa_in_b  = (const float*)d_in[3];
  const float* sa_out_w = (const float*)d_in[4];
  const float* sa_out_b = (const float*)d_in[5];
  const float* ln1_g    = (const float*)d_in[6];
  const float* ln1_b    = (const float*)d_in[7];
  const float* cq_w     = (const float*)d_in[8];
  const float* cq_b     = (const float*)d_in[9];
  const float* ck_w     = (const float*)d_in[10];
  const float* ck_b     = (const float*)d_in[11];
  const float* cv_w     = (const float*)d_in[12];
  const float* cv_b     = (const float*)d_in[13];
  const float* co_w     = (const float*)d_in[14];
  const float* co_b     = (const float*)d_in[15];
  const float* ln2_g    = (const float*)d_in[16];
  const float* ln2_b    = (const float*)d_in[17];
  const float* w1       = (const float*)d_in[18];
  const float* b1       = (const float*)d_in[19];
  const float* w2       = (const float*)d_in[20];
  const float* b2       = (const float*)d_in[21];
  const float* ln3_g    = (const float*)d_in[22];
  const float* ln3_b    = (const float*)d_in[23];

  const int M1 = BB * QL;   // 1024 query rows
  const int MS = BB * SL;   // 32768 context rows

  char* p = (char*)d_ws;
  auto alloc = [&](size_t bytes) -> void* {
    void* r = (void*)p;
    p += (bytes + 255) & ~(size_t)255;
    return r;
  };

  h16* w_sa_in_h  = (h16*)alloc((size_t)3 * D_MODEL * D_MODEL * sizeof(h16));
  h16* w_sa_out_h = (h16*)alloc((size_t)D_MODEL * D_MODEL * sizeof(h16));
  h16* w_cq_h     = (h16*)alloc((size_t)D_MODEL * D_MODEL * sizeof(h16));
  h16* w_ck_h     = (h16*)alloc((size_t)D_MODEL * D_MODEL * sizeof(h16));
  h16* w_cv_h     = (h16*)alloc((size_t)D_MODEL * D_MODEL * sizeof(h16));
  h16* w_co_h     = (h16*)alloc((size_t)D_MODEL * D_MODEL * sizeof(h16));
  h16* w1_h       = (h16*)alloc((size_t)DFF_ * D_MODEL * sizeof(h16));
  h16* w2_h       = (h16*)alloc((size_t)D_MODEL * DFF_ * sizeof(h16));
  h16* ctx_h      = (h16*)alloc((size_t)MS * D_MODEL * sizeof(h16));
  h16* ln_h       = (h16*)alloc((size_t)M1 * D_MODEL * sizeof(h16));
  h16* qk_h       = (h16*)alloc((size_t)M1 * 2 * D_MODEL * sizeof(h16));
  h16* vT_self    = (h16*)alloc((size_t)D_MODEL * M1 * sizeof(h16));   // [D][B*Q]
  h16* attn_h     = (h16*)alloc((size_t)M1 * D_MODEL * sizeof(h16));
  h16* cq_h       = (h16*)alloc((size_t)M1 * D_MODEL * sizeof(h16));
  h16* ck_h       = (h16*)alloc((size_t)MS * D_MODEL * sizeof(h16));
  h16* vT_cross   = (h16*)alloc((size_t)D_MODEL * MS * sizeof(h16));   // [D][B*S]
  h16* ffn_h      = (h16*)alloc((size_t)M1 * DFF_ * sizeof(h16));
  float* x1       = (float*)alloc((size_t)M1 * D_MODEL * sizeof(float));
  float* x2       = (float*)alloc((size_t)M1 * D_MODEL * sizeof(float));

  auto CONV = [&](const float* src, h16* dst, size_t n) {
    f32_to_f16_kernel<<<2048, 256, 0, stream>>>(src, dst, n);
  };
  auto LN = [&](const float* x, const float* g, const float* b, h16* out) {
    ln_f16_kernel<<<M1, 256, 0, stream>>>(x, g, b, out, D_MODEL);
  };
  auto grid = [](int M, int N) { return dim3((unsigned)(N / GBN), (unsigned)(M / GBM)); };

  // --- weight / context precision conversion ---
  CONV(sa_in_w,  w_sa_in_h,  (size_t)3 * D_MODEL * D_MODEL);
  CONV(sa_out_w, w_sa_out_h, (size_t)D_MODEL * D_MODEL);
  CONV(cq_w, w_cq_h, (size_t)D_MODEL * D_MODEL);
  CONV(ck_w, w_ck_h, (size_t)D_MODEL * D_MODEL);
  CONV(cv_w, w_cv_h, (size_t)D_MODEL * D_MODEL);
  CONV(co_w, w_co_h, (size_t)D_MODEL * D_MODEL);
  CONV(w1, w1_h, (size_t)DFF_ * D_MODEL);
  CONV(w2, w2_h, (size_t)D_MODEL * DFF_);
  CONV(context, ctx_h, (size_t)MS * D_MODEL);

  // GEMM variants: <DOGELU, HASRES, O32, O16, O16T>

  // --- self-attention block ---
  LN(queries, ln1_g, ln1_b, ln_h);
  // Q,K projection (first 2*D rows of combined in-proj weight)
  gemm_wmma_f16<0, 0, 0, 1, 0><<<grid(M1, 2 * D_MODEL), 256, 0, stream>>>(
      ln_h, w_sa_in_h, sa_in_b, nullptr, nullptr, qk_h, M1, 2 * D_MODEL, D_MODEL);
  // V projection, written transposed: vT_self[d][b*Q+s]
  gemm_wmma_f16<0, 0, 0, 0, 1><<<grid(M1, D_MODEL), 256, 0, stream>>>(
      ln_h, w_sa_in_h + (size_t)2 * D_MODEL * D_MODEL, sa_in_b + 2 * D_MODEL,
      nullptr, nullptr, vT_self, M1, D_MODEL, D_MODEL);
  attn_wmma<<<BB * NHEAD * (QL / 64), 256, 0, stream>>>(
      qk_h, qk_h, vT_self, attn_h, QL, QL,
      2 * D_MODEL, 2 * D_MODEL, M1, 0, D_MODEL);
  gemm_wmma_f16<0, 1, 1, 0, 0><<<grid(M1, D_MODEL), 256, 0, stream>>>(
      attn_h, w_sa_out_h, sa_out_b, queries, x1, nullptr, M1, D_MODEL, D_MODEL);

  // --- cross-attention block ---
  LN(x1, ln2_g, ln2_b, ln_h);
  gemm_wmma_f16<0, 0, 0, 1, 0><<<grid(M1, D_MODEL), 256, 0, stream>>>(
      ln_h, w_cq_h, cq_b, nullptr, nullptr, cq_h, M1, D_MODEL, D_MODEL);
  gemm_wmma_f16<0, 0, 0, 1, 0><<<grid(MS, D_MODEL), 256, 0, stream>>>(
      ctx_h, w_ck_h, ck_b, nullptr, nullptr, ck_h, MS, D_MODEL, D_MODEL);
  // V projection of context, written transposed: vT_cross[d][b*S+s]
  gemm_wmma_f16<0, 0, 0, 0, 1><<<grid(MS, D_MODEL), 256, 0, stream>>>(
      ctx_h, w_cv_h, cv_b, nullptr, nullptr, vT_cross, MS, D_MODEL, D_MODEL);
  attn_wmma<<<BB * NHEAD * (QL / 64), 256, 0, stream>>>(
      cq_h, ck_h, vT_cross, attn_h, QL, SL,
      D_MODEL, D_MODEL, MS, 0, 0);
  gemm_wmma_f16<0, 1, 1, 0, 0><<<grid(M1, D_MODEL), 256, 0, stream>>>(
      attn_h, w_co_h, co_b, x1, x2, nullptr, M1, D_MODEL, D_MODEL);

  // --- FFN block ---
  LN(x2, ln3_g, ln3_b, ln_h);
  gemm_wmma_f16<1, 0, 0, 1, 0><<<grid(M1, DFF_), 256, 0, stream>>>(
      ln_h, w1_h, b1, nullptr, nullptr, ffn_h, M1, DFF_, D_MODEL);
  gemm_wmma_f16<0, 1, 1, 0, 0><<<grid(M1, D_MODEL), 256, 0, stream>>>(
      ffn_h, w2_h, b2, x2, (float*)d_out, nullptr, M1, D_MODEL, DFF_);
}
